// H_Encoder_seg_2568390443074
// MI455X (gfx1250) — compile-verified, matrix operand order checked by reading, not run on hardware
//
#include <hip/hip_runtime.h>
#include <math.h>

// ---------------------------------------------------------------------------
// CDNA5 (gfx1250): all GEMMs via v_wmma_f32_16x16x32_f16, f32 activations in
// HBM/L2, f16 conversion at fragment load, f32 accumulation. wave32.
// Block = 128 threads = 4 waves tiling 32(M) x 64(N); each wave: 16x32 via
// two accumulators sharing one A fragment. Main K-loop is branchless
// (clamped rows/cols, float4 loads); only the K-tail uses select-zero.
// ---------------------------------------------------------------------------

typedef __attribute__((ext_vector_type(16))) _Float16 v16h;
typedef __attribute__((ext_vector_type(8)))  float    v8f;
typedef __attribute__((ext_vector_type(4)))  float    v4f;

#define NEGBIG (-3.4e38f)

// ---- WMMA fragment loaders (ISA 7.12.2 16-bit layout) ---------------------
// lane<16 holds K {0..7,16..23}, lane>=16 holds K {8..15,24..31} of its row.
// Row index is CLAMPED (safe: garbage rows only feed store-guarded outputs).

template <int VEC>
__device__ __forceinline__ v16h frag_row_main(const float* __restrict__ P, long ld,
                                              int rows, int r0, int k0, int lane) {
  int r = lane & 15, half = lane >> 4;
  int row = r0 + r; row = row < rows ? row : rows - 1;
  const float* p = P + (long)row * ld + k0 + half * 8;
  v16h f;
  if (VEC) {
    const v4f* q0 = (const v4f*)p;
    const v4f* q1 = (const v4f*)(p + 16);
    v4f a0 = q0[0], a1 = q0[1], b0 = q1[0], b1 = q1[1];
#pragma unroll
    for (int j = 0; j < 4; ++j) {
      f[j]      = (_Float16)a0[j];
      f[j + 4]  = (_Float16)a1[j];
      f[j + 8]  = (_Float16)b0[j];
      f[j + 12] = (_Float16)b1[j];
    }
  } else {
#pragma unroll
    for (int j = 0; j < 8; ++j) {
      f[j]     = (_Float16)p[j];
      f[j + 8] = (_Float16)p[16 + j];
    }
  }
  return f;
}

// K-tail: unconditional in-bounds load (k clamped) + select-zero, no branches.
__device__ __forceinline__ v16h frag_row_tail(const float* __restrict__ P, long ld,
                                              int rows, int r0, int k0, int K, int lane) {
  int r = lane & 15, half = lane >> 4;
  int row = r0 + r; row = row < rows ? row : rows - 1;
  const float* p = P + (long)row * ld;
  v16h f;
#pragma unroll
  for (int j = 0; j < 8; ++j) {
    int ka = k0 + half * 8 + j;
    int kb = ka + 16;
    float x0 = p[ka < K ? ka : 0];
    float x1 = p[kb < K ? kb : 0];
    f[j]     = (_Float16)(ka < K ? x0 : 0.0f);
    f[j + 8] = (_Float16)(kb < K ? x1 : 0.0f);
  }
  return f;
}

// B from plain [K, N] row-major (probs @ V): lane owns column n0+r (clamped),
// gathers the K pattern down the column (stride-ld scalar loads).
__device__ __forceinline__ v16h frag_col_main(const float* __restrict__ P, long ld,
                                              int ncols, int n0, int k0, int lane) {
  int r = lane & 15, half = lane >> 4;
  int col = n0 + r; col = col < ncols ? col : ncols - 1;
  const float* p = P + (long)(k0 + half * 8) * ld + col;
  v16h f;
#pragma unroll
  for (int j = 0; j < 8; ++j) {
    f[j]     = (_Float16)p[(long)j * ld];
    f[j + 8] = (_Float16)p[(long)(16 + j) * ld];
  }
  return f;
}

__device__ __forceinline__ v16h frag_col_tail(const float* __restrict__ P, long ld,
                                              int ncols, int n0, int k0, int K, int lane) {
  int r = lane & 15, half = lane >> 4;
  int col = n0 + r; col = col < ncols ? col : ncols - 1;
  v16h f;
#pragma unroll
  for (int j = 0; j < 8; ++j) {
    int ka = k0 + half * 8 + j;
    int kb = ka + 16;
    float x0 = P[(long)(ka < K ? ka : 0) * ld + col];
    float x1 = P[(long)(kb < K ? kb : 0) * ld + col];
    f[j]     = (_Float16)(ka < K ? x0 : 0.0f);
    f[j + 8] = (_Float16)(kb < K ? x1 : 0.0f);
  }
  return f;
}

// ---- epilogue: one 16x16 f32 accumulator -> C with fused bias/act/residual
template <int ACT, int HASBIAS, int HASRES>
__device__ __forceinline__ void epilogue16(float* __restrict__ C, const float* __restrict__ bias,
                                           const float* __restrict__ Rr, long ldc,
                                           int M, int N, int m0, int n0, int lane, v8f acc) {
  if (n0 >= N) return;                       // wave-uniform
  int r = lane & 15, half = lane >> 4;
  int n = n0 + r;
  bool nok = n < N;
  float bv = (HASBIAS && nok) ? bias[n] : 0.0f;
#pragma unroll
  for (int i = 0; i < 8; ++i) {              // VGPR i -> row m0 + 8*half + i
    int m = m0 + half * 8 + i;
    if (m >= M || !nok) continue;
    float v = acc[i] + bv;
    if (ACT == 1) v = fmaxf(v, 0.0f);
    else if (ACT == 2) v = 0.5f * v * (1.0f + erff(v * 0.70710678118654752f));
    if (HASRES) v += Rr[(long)m * ldc + n];
    C[(long)m * ldc + n] = v;
  }
}

// ---- Generic batched WMMA GEMM --------------------------------------------
// C[z] = act(A[z] (MxK) * B[z] + bias) [+ R[z]]
// BMODE 0: B source [N,K] row-major (x @ W^T); BMODE 1: B source [K,N].
// VEC 1: lda/ldb %4==0 -> float4 main-path loads.  z -> (b=z/H, h=z%H).
template <int BMODE, int ACT, int HASBIAS, int HASRES, int VEC>
__global__ void gemm_wmma_kernel(const float* __restrict__ A, const float* __restrict__ Bm,
                                 const float* __restrict__ bias, const float* __restrict__ Rr,
                                 float* __restrict__ C,
                                 int M, int N, int K,
                                 long lda, long ldb, long ldc, int H,
                                 long sAb, long sAh, long sBb, long sBh,
                                 long sCb, long sCh, long sRb, long sRh) {
  int lane = threadIdx.x & 31;
  int wave = threadIdx.x >> 5;
  int m0 = blockIdx.y * 32 + (wave >> 1) * 16;
  int n0 = blockIdx.x * 64 + (wave & 1) * 32;      // two 16-wide subtiles
  int z = blockIdx.z;
  int bb = z / H, hh = z % H;
  A  += (long)bb * sAb + (long)hh * sAh;
  Bm += (long)bb * sBb + (long)hh * sBh;
  C  += (long)bb * sCb + (long)hh * sCh;
  if (HASRES) Rr += (long)bb * sRb + (long)hh * sRh;
  if (m0 >= M || n0 >= N) return;                  // wave-uniform exit

#define LOADB_MAIN(kk, nn) ((BMODE == 0) ? frag_row_main<VEC>(Bm, ldb, N, (nn), (kk), lane) \
                                         : frag_col_main(Bm, ldb, N, (nn), (kk), lane))
#define LOADB_TAIL(kk, nn) ((BMODE == 0) ? frag_row_tail(Bm, ldb, N, (nn), (kk), K, lane) \
                                         : frag_col_tail(Bm, ldb, N, (nn), (kk), K, lane))

  v8f acc0 = {0.f, 0.f, 0.f, 0.f, 0.f, 0.f, 0.f, 0.f};
  v8f acc1 = acc0;
  int kmain = K & ~31;
  int k0 = 0;
  for (; k0 + 64 <= kmain; k0 += 64) {             // unroll x2: overlap loads/WMMA
    v16h a0  = frag_row_main<VEC>(A, lda, M, m0, k0, lane);
    v16h b00 = LOADB_MAIN(k0, n0);
    v16h b01 = LOADB_MAIN(k0, n0 + 16);
    v16h a1  = frag_row_main<VEC>(A, lda, M, m0, k0 + 32, lane);
    v16h b10 = LOADB_MAIN(k0 + 32, n0);
    v16h b11 = LOADB_MAIN(k0 + 32, n0 + 16);
    acc0 = __builtin_amdgcn_wmma_f32_16x16x32_f16(false, a0, false, b00, (short)0, acc0, false, false);
    acc1 = __builtin_amdgcn_wmma_f32_16x16x32_f16(false, a0, false, b01, (short)0, acc1, false, false);
    acc0 = __builtin_amdgcn_wmma_f32_16x16x32_f16(false, a1, false, b10, (short)0, acc0, false, false);
    acc1 = __builtin_amdgcn_wmma_f32_16x16x32_f16(false, a1, false, b11, (short)0, acc1, false, false);
  }
  for (; k0 < kmain; k0 += 32) {
    v16h a  = frag_row_main<VEC>(A, lda, M, m0, k0, lane);
    v16h b0 = LOADB_MAIN(k0, n0);
    v16h b1 = LOADB_MAIN(k0, n0 + 16);
    acc0 = __builtin_amdgcn_wmma_f32_16x16x32_f16(false, a, false, b0, (short)0, acc0, false, false);
    acc1 = __builtin_amdgcn_wmma_f32_16x16x32_f16(false, a, false, b1, (short)0, acc1, false, false);
  }
  if (k0 < K) {                                    // branchless zero-padded tail
    v16h a  = frag_row_tail(A, lda, M, m0, k0, K, lane);
    v16h b0 = LOADB_TAIL(k0, n0);
    v16h b1 = LOADB_TAIL(k0, n0 + 16);
    acc0 = __builtin_amdgcn_wmma_f32_16x16x32_f16(false, a, false, b0, (short)0, acc0, false, false);
    acc1 = __builtin_amdgcn_wmma_f32_16x16x32_f16(false, a, false, b1, (short)0, acc1, false, false);
  }
#undef LOADB_MAIN
#undef LOADB_TAIL

  epilogue16<ACT, HASBIAS, HASRES>(C, bias, Rr, ldc, M, N, m0, n0, lane, acc0);
  epilogue16<ACT, HASBIAS, HASRES>(C, bias, Rr, ldc, M, N, m0, n0 + 16, lane, acc1);
}

// ---- small fused/elementwise kernels --------------------------------------

__global__ void bn_fold_kernel(const float* __restrict__ W, const float* __restrict__ cb,
                               const float* __restrict__ g, const float* __restrict__ beta,
                               const float* __restrict__ mean, const float* __restrict__ var,
                               float* __restrict__ Wo, float* __restrict__ bo, int N, int K) {
  long t = (long)blockIdx.x * blockDim.x + threadIdx.x;
  if (t >= (long)N * K) return;
  int n = (int)(t / K), k = (int)(t % K);
  float s = g[n] * rsqrtf(var[n] + 1e-5f);
  Wo[t] = W[t] * s;
  if (k == 0) bo[n] = (cb[n] - mean[n]) * s + beta[n];
}

__global__ void rowmax_kernel(const float* __restrict__ X, float* __restrict__ Y,
                              long R, int Kp, int C) {
  long t = (long)blockIdx.x * blockDim.x + threadIdx.x;
  if (t >= R * C) return;
  long r = t / C; int c = (int)(t % C);
  const float* p = X + (r * (long)Kp) * C + c;
  float m = p[0];
  for (int kk = 1; kk < Kp; ++kk) m = fmaxf(m, p[(long)kk * C]);
  Y[r * (long)C + c] = m;
}

__global__ void concat_kernel(const float* __restrict__ gmax, const float* __restrict__ f,
                              float* __restrict__ out, long R, int Kp, int C) {
  long t = (long)blockIdx.x * blockDim.x + threadIdx.x;
  long total = R * (long)Kp * (2 * C);
  if (t >= total) return;
  long row = t / (2 * C); int c = (int)(t % (2 * C));
  long r = row / Kp;
  out[t] = (c < C) ? gmax[r * (long)C + c] : f[row * (long)C + (c - C)];
}

__global__ void add_kernel(const float* __restrict__ a, const float* __restrict__ b,
                           float* __restrict__ c, long n) {
  long t = (long)blockIdx.x * blockDim.x + threadIdx.x;
  if (t < n) c[t] = a[t] + b[t];
}

__global__ void gather_kernel(const float* __restrict__ X, const int* __restrict__ idx,
                              float* __restrict__ Y, long rows, int C) {
  long t = (long)blockIdx.x * blockDim.x + threadIdx.x;
  if (t >= rows * C) return;
  long r = t / C; int c = (int)(t % C);
  Y[t] = X[(long)idx[r] * C + c];
}

__global__ void layernorm_kernel(const float* __restrict__ X, const float* __restrict__ g,
                                 const float* __restrict__ b, float* __restrict__ Y, int C) {
  long row = blockIdx.x;
  const float* x = X + row * (long)C;
  float s = 0.f, ss = 0.f;
  for (int c = threadIdx.x; c < C; c += 32) { float v = x[c]; s += v; ss += v * v; }
#pragma unroll
  for (int o = 16; o > 0; o >>= 1) { s += __shfl_xor(s, o, 32); ss += __shfl_xor(ss, o, 32); }
  float mu = s / C, var = ss / C - mu * mu;
  float inv = rsqrtf(var + 1e-5f);
  float* y = Y + row * (long)C;
  for (int c = threadIdx.x; c < C; c += 32) y[c] = (x[c] - mu) * inv * g[c] + b[c];
}

__global__ void ln_transpose_kernel(const float* __restrict__ X, const float* __restrict__ g,
                                    const float* __restrict__ b, float* __restrict__ Y,
                                    int N, int C) {
  long row = blockIdx.x;
  long bb = row / N; int nn = (int)(row % N);
  const float* x = X + row * (long)C;
  float s = 0.f, ss = 0.f;
  for (int c = threadIdx.x; c < C; c += 32) { float v = x[c]; s += v; ss += v * v; }
#pragma unroll
  for (int o = 16; o > 0; o >>= 1) { s += __shfl_xor(s, o, 32); ss += __shfl_xor(ss, o, 32); }
  float mu = s / C, var = ss / C - mu * mu;
  float inv = rsqrtf(var + 1e-5f);
  float* y = Y + bb * (long)C * N + nn;
  for (int c = threadIdx.x; c < C; c += 32) y[(long)c * N] = (x[c] - mu) * inv * g[c] + b[c];
}

__global__ void softmax_mask_kernel(float* __restrict__ S, const float* __restrict__ ctr,
                                    int H, int N, float scale, float r2) {
  int i = blockIdx.x;
  int z = blockIdx.y;
  int bb = z / H;
  float* row = S + ((long)z * N + i) * N;
  const float* ci = ctr + ((long)bb * N + i) * 3;
  float cx = ci[0], cy = ci[1], cz = ci[2];
  int lane = threadIdx.x;
  float vals[16];
  int cnt = 0;
  float mx = NEGBIG;
  for (int j = lane; j < N; j += 32) {
    const float* cj = ctr + ((long)bb * N + j) * 3;
    float dx = cx - cj[0], dy = cy - cj[1], dz = cz - cj[2];
    float d2 = dx * dx + dy * dy + dz * dz;
    float v = (d2 >= r2) ? NEGBIG : row[j] * scale;
    vals[cnt++] = v;
    mx = fmaxf(mx, v);
  }
#pragma unroll
  for (int o = 16; o > 0; o >>= 1) mx = fmaxf(mx, __shfl_xor(mx, o, 32));
  float sum = 0.f; cnt = 0;
  for (int j = lane; j < N; j += 32) { float e = __expf(vals[cnt] - mx); vals[cnt++] = e; sum += e; }
#pragma unroll
  for (int o = 16; o > 0; o >>= 1) sum += __shfl_xor(sum, o, 32);
  float inv = 1.0f / sum;
  cnt = 0;
  for (int j = lane; j < N; j += 32) row[j] = vals[cnt++] * inv;
}

// ---- host-side GEMM launcher ----------------------------------------------
template <int BM, int AC, int BI, int RE>
static void gemmL(hipStream_t stream, const float* A, const float* B, const float* bias,
                  const float* R, float* C, int M, int N, int K,
                  long lda, long ldb, long ldc,
                  int batch = 1, int H = 1,
                  long sAb = 0, long sAh = 0, long sBb = 0, long sBh = 0,
                  long sCb = 0, long sCh = 0, long sRb = 0, long sRh = 0) {
  dim3 g((unsigned)((N + 63) / 64), (unsigned)((M + 31) / 32), (unsigned)batch);
  bool vec = ((lda & 3) == 0) && (BM == 1 || ((ldb & 3) == 0));
  if (vec)
    gemm_wmma_kernel<BM, AC, BI, RE, 1><<<g, 128, 0, stream>>>(
        A, B, bias, R, C, M, N, K, lda, ldb, ldc, H, sAb, sAh, sBb, sBh, sCb, sCh, sRb, sRh);
  else
    gemm_wmma_kernel<BM, AC, BI, RE, 0><<<g, 128, 0, stream>>>(
        A, B, bias, R, C, M, N, K, lda, ldb, ldc, H, sAb, sAh, sBb, sBh, sCb, sCh, sRb, sRh);
}

static inline unsigned ewb(long n) { return (unsigned)((n + 255) / 256); }

// ---------------------------------------------------------------------------
extern "C" void kernel_launch(void* const* d_in, const int* in_sizes, int n_in,
                              void* d_out, int out_size, void* d_ws, size_t ws_size,
                              hipStream_t stream) {
  (void)in_sizes; (void)n_in; (void)out_size; (void)ws_size;

  const int   Bsz     = 32;
  const int   Gs[3]   = {512, 256, 64};
  const int   Kp_[3]  = {16, 8, 8};
  const int   Cin_[3] = {3, 96, 192};
  const int   Cd_[3]  = {96, 192, 384};
  const float RAD2[3] = {0.32f * 0.32f, 0.64f * 0.64f, 1.28f * 1.28f};

  // param leaves in JAX pytree order (dict keys sorted, lists in order)
  auto blk = [&](int st, int l, int f) { return (const float*)d_in[(st * 5 + l) * 11 + f]; };
  auto nrm = [&](int st, int f) { return (const float*)d_in[165 + st * 2 + f]; };
  auto pw  = [&](int st, int f) { return (const float*)d_in[171 + st * 4 + f]; };
  auto tok = [&](int st, int f) { return (const float*)d_in[183 + st * 16 + f]; };
  const float* neigh0    = (const float*)d_in[231];
  const float* center[3] = {(const float*)d_in[234], (const float*)d_in[235],
                            (const float*)d_in[236]};
  const int* idxs[2] = {(const int*)d_in[237], (const int*)d_in[238]};

  // workspace carve (~1.6 GB peak; buffers aliased by liveness)
  char* wsb = (char*)d_ws;
  size_t off = 0;
  auto wsAlloc = [&](size_t bytes) -> float* {
    float* p = (float*)(wsb + off);
    off += (bytes + 255) & ~(size_t)255;
    return p;
  };
  float* bufCat  = wsAlloc(536870912ull);
  float* bufE    = wsAlloc(536870912ull);
  float* bufF2   = wsAlloc(268435456ull);   // f2; reused as attention scores
  float* bufE4   = wsAlloc(100663296ull);
  float* bufGmax = wsAlloc(16777216ull);
  float* bufGat  = wsAlloc(33554432ull);
  float* xvis[3];
  xvis[0] = wsAlloc((size_t)Bsz * 512 * 96 * 4);
  xvis[1] = wsAlloc((size_t)Bsz * 256 * 192 * 4);
  xvis[2] = wsAlloc((size_t)Bsz * 64 * 384 * 4);
  float* bufXin = wsAlloc(6291456);
  float* bufH   = wsAlloc(6291456);
  float* bufO   = wsAlloc(6291456);
  float* bufPos = wsAlloc(6291456);
  float* bufPt  = wsAlloc(6291456);
  float* bufQKV = wsAlloc(18874368);
  float* bufFFN = wsAlloc(25165824);
  float* W1f = wsAlloc(128 * 192 * 4);
  float* b1f = wsAlloc(128 * 4);
  float* W3f = wsAlloc(512 * 512 * 4);
  float* b3f = wsAlloc(512 * 4);
  float* bufS = bufF2;

  for (int st = 0; st < 3; ++st) {
    const int G = Gs[st], Kp = Kp_[st], Ci = Cin_[st], C = Cd_[st];
    const long Rg = (long)Bsz * G;
    const long Rows = Rg * Kp;
    const int H = 6, hd = C / 6, Ntok = G;
    float* X = xvis[st];

    // ---- token embed -------------------------------------------------------
    const float* Ain;
    if (st == 0) {
      Ain = neigh0;
    } else {
      long n = Rows * Ci;
      gather_kernel<<<ewb(n), 256, 0, stream>>>(xvis[st - 1], idxs[st - 1], bufGat, Rows, Ci);
      Ain = bufGat;
    }
    { long n = (long)128 * Ci;
      bn_fold_kernel<<<ewb(n), 256, 0, stream>>>(tok(st, 9), tok(st, 8), tok(st, 1),
                                                 tok(st, 0), tok(st, 2), tok(st, 3),
                                                 W1f, b1f, 128, Ci); }
    gemmL<0, 1, 1, 0>(stream, Ain, W1f, b1f, nullptr, bufE, (int)Rows, 128, Ci, Ci, Ci, 128);
    gemmL<0, 0, 1, 0>(stream, bufE, tok(st, 11), tok(st, 10), nullptr, bufF2,
                      (int)Rows, 256, 128, 128, 128, 256);
    { long n = Rg * 256;
      rowmax_kernel<<<ewb(n), 256, 0, stream>>>(bufF2, bufGmax, Rg, Kp, 256); }
    { long n = Rows * 512;
      concat_kernel<<<ewb(n), 256, 0, stream>>>(bufGmax, bufF2, bufCat, Rg, Kp, 256); }
    { long n = (long)512 * 512;
      bn_fold_kernel<<<ewb(n), 256, 0, stream>>>(tok(st, 13), tok(st, 12), tok(st, 5),
                                                 tok(st, 4), tok(st, 6), tok(st, 7),
                                                 W3f, b3f, 512, 512); }
    gemmL<0, 1, 1, 0>(stream, bufCat, W3f, b3f, nullptr, bufE, (int)Rows, 512, 512, 512, 512, 512);
    gemmL<0, 0, 1, 0>(stream, bufE, tok(st, 15), tok(st, 14), nullptr, bufE4,
                      (int)Rows, C, 512, 512, 512, C);
    { long n = Rg * C;
      rowmax_kernel<<<ewb(n), 256, 0, stream>>>(bufE4, X, Rg, Kp, C); }

    // ---- positional MLP ----------------------------------------------------
    gemmL<0, 2, 1, 0>(stream, center[st], pw(st, 2), pw(st, 0), nullptr, bufPt,
                      (int)Rg, C, 3, 3, 3, C);
    gemmL<0, 0, 1, 0>(stream, bufPt, pw(st, 3), pw(st, 1), nullptr, bufPos,
                      (int)Rg, C, C, C, C, C);

    // ---- transformer blocks ------------------------------------------------
    const float scale = 1.0f / sqrtf((float)hd);
    for (int l = 0; l < 5; ++l) {
      long nEl = Rg * C;
      add_kernel<<<ewb(nEl), 256, 0, stream>>>(X, bufPos, bufXin, nEl);
      layernorm_kernel<<<(unsigned)Rg, 32, 0, stream>>>(bufXin, blk(st, l, 5),
                                                        blk(st, l, 4), bufH, C);
      gemmL<0, 0, 0, 0>(stream, bufH, blk(st, l, 10), nullptr, nullptr, bufQKV,
                        (int)Rg, 3 * C, C, C, C, 3 * C);
      // scores[z=b*H+h] = Q K^T  (Kdim = hd; hd=16 handled by zero-padded tail)
      gemmL<0, 0, 0, 0>(stream, bufQKV, bufQKV + C, nullptr, nullptr, bufS,
                        Ntok, Ntok, hd, 3 * C, 3 * C, Ntok,
                        Bsz * H, H,
                        (long)Ntok * 3 * C, (long)hd,
                        (long)Ntok * 3 * C, (long)hd,
                        (long)H * Ntok * Ntok, (long)Ntok * Ntok, 0, 0);
      softmax_mask_kernel<<<dim3((unsigned)Ntok, (unsigned)(Bsz * H)), 32, 0, stream>>>(
          bufS, center[st], H, Ntok, scale, RAD2[st]);
      // O = probs @ V
      gemmL<1, 0, 0, 0>(stream, bufS, bufQKV + 2 * C, nullptr, nullptr, bufO,
                        Ntok, hd, Ntok, Ntok, 3 * C, C,
                        Bsz * H, H,
                        (long)H * Ntok * Ntok, (long)Ntok * Ntok,
                        (long)Ntok * 3 * C, (long)hd,
                        (long)Ntok * C, (long)hd, 0, 0);
      // x = xin + proj(O)
      gemmL<0, 0, 1, 1>(stream, bufO, blk(st, l, 9), blk(st, l, 8), bufXin, X,
                        (int)Rg, C, C, C, C, C);
      layernorm_kernel<<<(unsigned)Rg, 32, 0, stream>>>(X, blk(st, l, 7),
                                                        blk(st, l, 6), bufH, C);
      gemmL<0, 2, 1, 0>(stream, bufH, blk(st, l, 1), blk(st, l, 0), nullptr, bufFFN,
                        (int)Rg, 4 * C, C, C, C, 4 * C);
      gemmL<0, 0, 1, 1>(stream, bufFFN, blk(st, l, 3), blk(st, l, 2), X, X,
                        (int)Rg, C, 4 * C, 4 * C, 4 * C, C);
    }
  }

  // ---- final LayerNorm + transpose into d_out ------------------------------
  float* out = (float*)d_out;
  size_t oo = 0;
  for (int st = 0; st < 3; ++st) {
    ln_transpose_kernel<<<(unsigned)(Bsz * Gs[st]), 32, 0, stream>>>(
        xvis[st], nrm(st, 1), nrm(st, 0), out + oo, Gs[st], Cd_[st]);
    oo += (size_t)Bsz * Cd_[st] * Gs[st];
  }
}
